// SoftAttention_34076270526485
// MI455X (gfx1250) — matrix operand order repaired and checked by reading, not compiled, not run on hardware
//
#include <hip/hip_runtime.h>

typedef __bf16 bf16_t;
typedef __attribute__((ext_vector_type(16))) __bf16 v16bf;
typedef __attribute__((ext_vector_type(8)))  float  v8f;
typedef __attribute__((ext_vector_type(4)))  int    i32x4;
typedef __attribute__((ext_vector_type(8)))  int    i32x8;
typedef __attribute__((ext_vector_type(4)))  unsigned int u32x4;

namespace {
constexpr int Ldim = 8192;
constexpr int Rdim = 8192;
constexpr int Ddim = 128;
constexpr int BR   = 64;    // Q rows per workgroup
constexpr int BL   = 64;    // K/V rows per L-tile
constexpr int NW   = 4;     // waves per workgroup (wave32)
constexpr int NT   = NW * 32;
// padded LDS strides: 16B-aligned rows, break the 256B bank wrap
constexpr int QS_STR = Ddim + 8;  // 136 elems = 272 B  (TDM pad: 64dw interval + 4dw pad)
constexpr int KS_STR = Ddim + 8;  // 136
constexpr int VT_STR = BL + 8;    // 72 elems = 144 B   (TDM pad: 32dw interval + 4dw pad)
constexpr int PS_STR = BL + 8;    // 72
constexpr float LOG2E = 1.4426950408889634f;
}

// fast device exp2 (raw v_exp_f32)
__device__ __forceinline__ float exp2_fast(float x) {
  return __builtin_amdgcn_exp2f(x);
}

// generic (flat) shared pointer -> 32-bit LDS byte offset for the TDM D#
typedef __attribute__((address_space(3))) void lds_void;
__device__ __forceinline__ unsigned lds_addr_of(void* p) {
  return (unsigned)(unsigned long long)(lds_void*)p;
}

// Issue a 2D TENSOR_LOAD_TO_LDS: tile (tile0 x tile1) elements of 2 bytes,
// row stride stride0 elements, LDS destination padded per g1w0 bits.
__device__ __forceinline__ void tdm_load_2d(unsigned lds_off,
                                            unsigned long long gaddr,
                                            unsigned g1w0,
                                            unsigned td0, unsigned td1,
                                            unsigned tile0, unsigned tile1,
                                            unsigned long long stride0) {
  u32x4 g0;
  g0[0] = 1u;                                   // count=1 (valid user descriptor)
  g0[1] = lds_off;                              // lds_addr [63:32]
  g0[2] = (unsigned)(gaddr & 0xffffffffull);    // global_addr lo
  g0[3] = (unsigned)((gaddr >> 32) & 0x1ffffffull) | (2u << 30); // addr hi | type=2
  i32x8 g1;
  g1[0] = (int)g1w0;                            // wg_mask=0 | data_size | pad ctl
  g1[1] = (int)((td0 & 0xffffu) << 16);         // tensor_dim0 [79:48]
  g1[2] = (int)(((td0 >> 16) & 0xffffu) | ((td1 & 0xffffu) << 16)); // td0 hi | td1 lo
  g1[3] = (int)(((td1 >> 16) & 0xffffu) | ((tile0 & 0xffffu) << 16)); // td1 hi | tile_dim0
  g1[4] = (int)(tile1 & 0xffffu);               // tile_dim1 | tile_dim2=0
  g1[5] = (int)(unsigned)(stride0 & 0xffffffffull);   // tensor_dim0_stride lo
  g1[6] = (int)(unsigned)((stride0 >> 32) & 0xffffu); // stride0 hi | stride1 lo=0
  g1[7] = 0;
  i32x4 zz = {0, 0, 0, 0};
#if __clang_major__ >= 23
  i32x8 z8 = {0, 0, 0, 0, 0, 0, 0, 0};
  __builtin_amdgcn_tensor_load_to_lds(g0, g1, zz, zz, z8, 0);
#else
  __builtin_amdgcn_tensor_load_to_lds(g0, g1, zz, zz, 0);
#endif
}

// ---------------- prep: fp32 -> bf16 (row-major K, transposed V, scaled Q) --
__global__ __launch_bounds__(128) void prep_bf16(const float* __restrict__ lin,
                                                 const float* __restrict__ rin,
                                                 bf16_t* __restrict__ lin_bf,   // [L][D]
                                                 bf16_t* __restrict__ linT_bf,  // [D][L]
                                                 bf16_t* __restrict__ rin_bf) { // [R][D], pre-scaled by log2(e)
  const int r = blockIdx.x;     // 0..8191
  const int c = threadIdx.x;    // 0..127
  const size_t idx = (size_t)r * Ddim + c;
  bf16_t kv = (bf16_t)lin[idx];
  lin_bf[idx] = kv;
  linT_bf[(size_t)c * Ldim + r] = kv;
  rin_bf[idx] = (bf16_t)(rin[idx] * LOG2E);   // softmax done in exp2 domain
}

// ---------------- flash attention with TDM double-buffered K/V --------------
__global__ __launch_bounds__(NT) void soft_attn_flash(
    const bf16_t* __restrict__ lin_bf,    // K rows   [L][D]
    const bf16_t* __restrict__ linT_bf,   // V cols   [D][L]
    const bf16_t* __restrict__ rin_bf,    // Q rows   [R][D] (x log2e)
    float* __restrict__ out)              // [R][D] fp32
{
  __shared__ bf16_t Qs[BR][QS_STR];          // 17.0 KB
  __shared__ bf16_t Ks[2][BL][KS_STR];       // 34.0 KB (double buffer)
  __shared__ bf16_t Vt[2][Ddim][VT_STR];     // 36.0 KB (double buffer)
  __shared__ bf16_t Ps[NW][16][PS_STR];      //  9.0 KB

  const int tid  = threadIdx.x;
  const int wave = tid >> 5;
  const int lane = tid & 31;
  const int n    = lane & 15;   // N (or M) index within a 16-lane half
  const int h    = lane >> 4;   // half selector (K-split in A/B layouts)
  const int r0   = blockIdx.x * BR;

  // descriptor constants
  const unsigned kW0 = (1u << 16) | (1u << 20) | (5u << 22) | (3u << 25); // 2B, pad 64dw+4dw
  const unsigned vW0 = (1u << 16) | (1u << 20) | (4u << 22) | (3u << 25); // 2B, pad 32dw+4dw
  const unsigned long long kBase = (unsigned long long)lin_bf;
  const unsigned long long vBase = (unsigned long long)linT_bf;
  const int T = Ldim / BL;     // 128 tiles

  if (wave == 0) {             // prologue: stage tile 0 (overlaps Q staging)
    tdm_load_2d(lds_addr_of(&Ks[0][0][0]), kBase, kW0, Ddim, BL, Ddim, BL, Ddim);
    tdm_load_2d(lds_addr_of(&Vt[0][0][0]), vBase, vW0, BL, Ddim, BL, Ddim, Ldim);
  }

  // ---- Q tile: straight bf16 copy global -> padded LDS --------------------
  #pragma unroll
  for (int i = 0; i < (BR * Ddim) / (NT * 8); ++i) {   // 8 iterations of 16B
    int flat = (i * NT + tid) * 8;
    int row  = flat >> 7;
    int col  = flat & 127;
    i32x4 v = *(const i32x4*)(rin_bf + (size_t)(r0 + row) * Ddim + col);
    *(i32x4*)&Qs[row][col] = v;
  }
  __syncthreads();

  // ---- Q A-operands (16x32 bf16 layout), kept in registers ----------------
  v16bf qa[4];
  #pragma unroll
  for (int kb = 0; kb < 4; ++kb) {
    const int k0 = kb * 32;
    i32x4* d = (i32x4*)&qa[kb];
    const bf16_t* s = &Qs[wave * 16 + n][k0 + h * 8];
    d[0] = *(const i32x4*)(s);
    d[1] = *(const i32x4*)(s + 16);
  }

  // all-ones bf16 B operand: P x ones = row sums on the matrix unit
  v16bf onesb;
  {
    i32x4 c1 = {0x3F803F80, 0x3F803F80, 0x3F803F80, 0x3F803F80};
    i32x4* o = (i32x4*)&onesb;
    o[0] = c1; o[1] = c1;
  }

  const v8f zero8 = {0.f, 0.f, 0.f, 0.f, 0.f, 0.f, 0.f, 0.f};
  v8f   of[8];
  float mrow[8], lrow[8];
  #pragma unroll
  for (int g = 0; g < 8; ++g) { of[g] = zero8; mrow[g] = -__builtin_inff(); lrow[g] = 0.f; }

  for (int t = 0; t < T; ++t) {
    const int cur = t & 1;
    if (wave == 0) __builtin_amdgcn_s_wait_tensorcnt(0);  // tile t resident
    __syncthreads();   // publish tile t; also fences tile t-1 readers below

    if (wave == 0 && t + 1 < T) {   // stage tile t+1 into the other buffer
      const int nxt = cur ^ 1;
      const unsigned long long koff = (unsigned long long)(t + 1) * BL * Ddim * 2ull;
      const unsigned long long voff = (unsigned long long)(t + 1) * BL * 2ull;
      tdm_load_2d(lds_addr_of(&Ks[nxt][0][0]), kBase + koff, kW0, Ddim, BL, Ddim, BL, Ddim);
      tdm_load_2d(lds_addr_of(&Vt[nxt][0][0]), vBase + voff, vW0, BL, Ddim, BL, Ddim, Ldim);
    }

    // ---- S = Q * K^T : 16x64 per wave (scores already in log2 domain) -----
    v8f st[4];
    #pragma unroll
    for (int ct = 0; ct < 4; ++ct) {
      v8f acc = zero8;
      #pragma unroll
      for (int kb = 0; kb < 4; ++kb) {
        v16bf bop;
        i32x4* d = (i32x4*)&bop;
        const bf16_t* s = &Ks[cur][ct * 16 + n][kb * 32 + h * 8];
        d[0] = *(const i32x4*)(s);
        d[1] = *(const i32x4*)(s + 16);
        acc = __builtin_amdgcn_wmma_f32_16x16x32_bf16(
                  false, qa[kb], false, bop, (short)0, acc, false, false);
      }
      st[ct] = acc;
    }

    // ---- online softmax in exp2 domain (register row g -> row g + 8*h) ----
    v8f sclv;
    #pragma unroll
    for (int g = 0; g < 8; ++g) {
      float mx = fmaxf(fmaxf(st[0][g], st[1][g]), fmaxf(st[2][g], st[3][g]));
      #pragma unroll
      for (int off = 1; off < 16; off <<= 1) mx = fmaxf(mx, __shfl_xor(mx, off));
      float mnew = fmaxf(mrow[g], mx);
      sclv[g] = exp2_fast(mrow[g] - mnew);
      #pragma unroll
      for (int ct = 0; ct < 4; ++ct) st[ct][g] = exp2_fast(st[ct][g] - mnew);
      mrow[g] = mnew;
    }
    #pragma unroll
    for (int dt = 0; dt < 8; ++dt) of[dt] = of[dt] * sclv;  // v_pk_mul_f32 pairs

    // ---- P (bf16) -> per-wave LDS buffer (A-layout source) ----------------
    #pragma unroll
    for (int ct = 0; ct < 4; ++ct)
      #pragma unroll
      for (int g = 0; g < 8; ++g)
        Ps[wave][g + 8 * h][ct * 16 + n] = (bf16_t)st[ct][g];

    // ---- O += P * V, and row-sums = P * ones (on the matrix unit) ---------
    v8f rsacc = zero8;
    #pragma unroll
    for (int kp = 0; kp < 2; ++kp) {
      v16bf pa;
      i32x4* d = (i32x4*)&pa;
      const bf16_t* s = &Ps[wave][n][kp * 32 + h * 8];
      d[0] = *(const i32x4*)(s);
      d[1] = *(const i32x4*)(s + 16);
      rsacc = __builtin_amdgcn_wmma_f32_16x16x32_bf16(
                  false, pa, false, onesb, (short)0, rsacc, false, false);
      #pragma unroll
      for (int dt = 0; dt < 8; ++dt) {
        v16bf vb;
        i32x4* e = (i32x4*)&vb;
        const bf16_t* vs = &Vt[cur][dt * 16 + n][kp * 32 + h * 8];
        e[0] = *(const i32x4*)(vs);
        e[1] = *(const i32x4*)(vs + 16);
        of[dt] = __builtin_amdgcn_wmma_f32_16x16x32_bf16(
                     false, pa, false, vb, (short)0, of[dt], false, false);
      }
    }
    // lrow tracks the sum of bf16-rounded P (exactly what PV accumulated)
    #pragma unroll
    for (int g = 0; g < 8; ++g) lrow[g] = lrow[g] * sclv[g] + rsacc[g];
  }

  // ---- epilogue: divide by row sum, store fp32 ----------------------------
  #pragma unroll
  for (int g = 0; g < 8; ++g) {
    float inv = 1.0f / lrow[g];
    int row = r0 + wave * 16 + g + 8 * h;
    #pragma unroll
    for (int dt = 0; dt < 8; ++dt)
      out[(size_t)row * Ddim + dt * 16 + n] = of[dt][g] * inv;
  }
}

extern "C" void kernel_launch(void* const* d_in, const int* in_sizes, int n_in,
                              void* d_out, int out_size, void* d_ws, size_t ws_size,
                              hipStream_t stream) {
  const float* lin = (const float*)d_in[0];   // linput [8192,128]
  const float* rin = (const float*)d_in[1];   // rinput [8192,128]
  float* out = (float*)d_out;                 // [8192,128] fp32

  bf16_t* lin_bf  = (bf16_t*)d_ws;                       // 2 MB
  bf16_t* linT_bf = lin_bf + (size_t)Ldim * Ddim;        // 2 MB
  bf16_t* rin_bf  = linT_bf + (size_t)Ldim * Ddim;       // 2 MB

  hipLaunchKernelGGL(prep_bf16, dim3(Ldim), dim3(128), 0, stream,
                     lin, rin, lin_bf, linT_bf, rin_bf);
  hipLaunchKernelGGL(soft_attn_flash, dim3(Rdim / BR), dim3(NT), 0, stream,
                     lin_bf, linT_bf, rin_bf, out);
}
